// SequenceLSTMDecoder_71390946394600
// MI455X (gfx1250) — matrix operand order repaired and checked by reading, not compiled, not run on hardware
//
#include <hip/hip_runtime.h>
#include <hip/hip_bf16.h>

// ---------------------------------------------------------------------------
// Problem constants (match reference): B=32, T=64, H=1024, V=32000
// ---------------------------------------------------------------------------
#define BB   32
#define TT   64
#define HH   1024
#define VV   32000
#define G4H  (4 * HH)      // 4096
#define MROWS (BB * TT)    // 2048

// LDS A-tile row stride (bf16 elems): 32 data + 8 pad = 80 bytes/row.
// 16 fragment rows then start at distinct bank groups ((r*20)%64 all unique).
#define APAD 40

typedef __attribute__((ext_vector_type(16))) __bf16 v16bf;
typedef __attribute__((ext_vector_type(8)))  float  v8f;

union FragBF { v16bf v; float4 f4[2]; };

// A-fragment: 16x32 (MxK) bf16 from row-major tile with leading dim `ld`.
// Lane L: m = L&15, half = L>>4.
// elems 0..7  -> K = 8*half + 0..7
// elems 8..15 -> K = 16 + 8*half + 0..7
__device__ __forceinline__ v16bf load_a_frag_g(const __bf16* __restrict__ tile, int ld) {
    const int lane = threadIdx.x & 31;
    const int m  = lane & 15;
    const int hf = lane >> 4;
    const __bf16* p = tile + (size_t)m * ld + 8 * hf;
    FragBF u;
    u.f4[0] = *(const float4*)(p);
    u.f4[1] = *(const float4*)(p + 16);
    return u.v;
}

// Same fragment sourced from an LDS tile (row stride APAD elems) -> ds_load_b128.
__device__ __forceinline__ v16bf load_a_frag_lds(const __bf16* sm_tile) {
    const int lane = threadIdx.x & 31;
    const int m  = lane & 15;
    const int hf = lane >> 4;
    const __bf16* p = sm_tile + m * APAD + 8 * hf;
    FragBF u;
    u.f4[0] = *(const float4*)(p);
    u.f4[1] = *(const float4*)(p + 16);
    return u.v;
}

// B-fragment: 32x16 (KxN) bf16 sourced from row-major weights [N x ld] (B = W^T).
// Lane L: n = L&15, half = L>>4; elems 0..7 -> K = 8*half + 0..7, elems 8..15 -> K = 16 + 8*half + 0..7.
__device__ __forceinline__ v16bf load_b_frag(const __bf16* __restrict__ wrows, int ld) {
    const int lane = threadIdx.x & 31;
    const int n  = lane & 15;
    const int hf = lane >> 4;
    const __bf16* p = wrows + (size_t)n * ld + 8 * hf;
    FragBF u;
    u.f4[0] = *(const float4*)(p);
    u.f4[1] = *(const float4*)(p + 16);
    return u.v;
}

__device__ __forceinline__ v8f wmma_bf16(v16bf a, v16bf b, v8f c) {
    return __builtin_amdgcn_wmma_f32_16x16x32_bf16(
        /*neg_a=*/false, a, /*neg_b=*/false, b,
        /*c_mod=*/(short)0, c, /*reuse_a=*/false, /*reuse_b=*/false);
}

__device__ __forceinline__ float dev_sigmoid(float x) {
    return 1.0f / (1.0f + __expf(-x));
}

// Async DMA of one 16B segment: global -> LDS (ASYNCcnt-tracked).
__device__ __forceinline__ void async_copy_b128(__bf16* lds_dst, const __bf16* gsrc) {
    const unsigned      lds = (unsigned)(uintptr_t)lds_dst;     // low 32b of flat LDS addr
    const unsigned long long ga = (unsigned long long)(uintptr_t)gsrc;
    asm volatile("global_load_async_to_lds_b128 %0, %1, off"
                 :: "v"(lds), "v"(ga) : "memory");
}

__device__ __forceinline__ void wait_asynccnt0() {
    asm volatile("s_wait_asynccnt 0x0" ::: "memory");
}

// ---------------------------------------------------------------------------
// bf16 WMMA GEMM: C[M,N] = A[M,K] * B[N,K]^T + bias[N]
// Block = 256 threads = 8 waves (2 x 4). Wave tile = 32(M) x 64(N).
// Block tile = 64(M) x 256(N). Requires M%64==0, N%256==0, K%32==0.
// A tiles (64x32 bf16 = 4 KB) are staged into LDS with async DMA, double-
// buffered: each of the 256 threads moves one 16 B segment per K-step.
// ---------------------------------------------------------------------------
__global__ __launch_bounds__(256)
void gemm_bf16_wmma(const __bf16* __restrict__ A, const __bf16* __restrict__ B,
                    float* __restrict__ C, const float* __restrict__ bias,
                    int M, int N, int K) {
    __shared__ __attribute__((aligned(16))) __bf16 smA[2][64 * APAD];

    const int wave = threadIdx.x >> 5;
    const int lane = threadIdx.x & 31;
    const int wm = wave >> 2;          // 0..1
    const int wn = wave & 3;           // 0..3
    const int mblk = blockIdx.y * 64;
    const int m0w  = wm * 32;          // wave M origin inside block tile
    const int n0   = blockIdx.x * 256 + wn * 64;

    const int srow = threadIdx.x >> 2;       // 0..63 : A-tile row this thread stages
    const int sseg = (threadIdx.x & 3) * 8;  // 0,8,16,24 : bf16 elems (16B segment)
    const __bf16* Abase = A + (size_t)(mblk + srow) * K + sseg;

    // Prologue: stage k=0 tile into buffer 0.
    async_copy_b128(&smA[0][srow * APAD + sseg], Abase);

    v8f acc[2][4] = {};
    int p = 0;
    for (int k = 0; k < K; k += 32) {
        wait_asynccnt0();      // this wave's staged segments have landed in LDS
        __syncthreads();       // all waves done staging buf[p] AND done reading buf[p^1]

        if (k + 32 < K)
            async_copy_b128(&smA[p ^ 1][srow * APAD + sseg], Abase + k + 32);

        if (k + 128 < K)
            __builtin_prefetch(B + (size_t)n0 * K + k + 128, 0, 1);

        v16bf a[2], b[4];
#pragma unroll
        for (int i = 0; i < 2; ++i)
            a[i] = load_a_frag_lds(&smA[p][(m0w + 16 * i) * APAD]);
#pragma unroll
        for (int j = 0; j < 4; ++j)
            b[j] = load_b_frag(B + (size_t)(n0 + 16 * j) * K + k, K);
#pragma unroll
        for (int i = 0; i < 2; ++i)
#pragma unroll
            for (int j = 0; j < 4; ++j)
                acc[i][j] = wmma_bf16(a[i], b[j], acc[i][j]);
        p ^= 1;
    }

    const int hf  = lane >> 4;
    const int col = lane & 15;
#pragma unroll
    for (int i = 0; i < 2; ++i) {
#pragma unroll
        for (int j = 0; j < 4; ++j) {
            const int n  = n0 + 16 * j + col;
            const float bv = bias[n];
#pragma unroll
            for (int r = 0; r < 8; ++r) {
                const int m = mblk + m0w + 16 * i + 8 * hf + r;
                C[(size_t)m * N + n] = acc[i][j][r] + bv;
            }
        }
    }
}

// ---------------------------------------------------------------------------
// One LSTM time step. 64 blocks x 128 threads (4 waves).
// Block owns a 16-wide slice of H. Wave w computes gate w for all 32 batch
// rows via WMMA (K=1024), stages gate tiles in LDS, then the block applies
// the LSTM nonlinearity and writes h (bf16, double-buffered) and hs.
// ---------------------------------------------------------------------------
__global__ __launch_bounds__(128)
void lstm_step(const __bf16* __restrict__ h_prev,   // [32,1024] bf16
               const __bf16* __restrict__ Whh,      // [4096,1024] bf16
               const float*  __restrict__ gx,       // [2048,4096] f32
               float*        __restrict__ c_state,  // [32,1024] f32
               __bf16*       __restrict__ h_next,   // [32,1024] bf16
               __bf16*       __restrict__ hs,       // [2048,1024] bf16
               int t) {
    __shared__ float gbuf[4][BB][16];

    const int wave  = threadIdx.x >> 5;   // gate index: 0=i 1=f 2=g 3=o
    const int lane  = threadIdx.x & 31;
    const int ncol0 = blockIdx.x * 16;    // hidden-column tile origin

    v8f acc[2] = {};
    const __bf16* Brow = Whh + (size_t)(wave * HH + ncol0) * HH;

    for (int k = 0; k < HH; k += 32) {
        v16bf a0 = load_a_frag_g(h_prev + k, HH);                    // rows 0..15
        v16bf a1 = load_a_frag_g(h_prev + (size_t)16 * HH + k, HH);  // rows 16..31
        v16bf bf = load_b_frag(Brow + k, HH);
        acc[0] = wmma_bf16(a0, bf, acc[0]);
        acc[1] = wmma_bf16(a1, bf, acc[1]);
    }

    const int hf  = lane >> 4;
    const int col = lane & 15;
#pragma unroll
    for (int i = 0; i < 2; ++i)
#pragma unroll
        for (int r = 0; r < 8; ++r)
            gbuf[wave][16 * i + 8 * hf + r][col] = acc[i][r];

    __syncthreads();

    // 32 batch rows x 16 cols = 512 gate columns handled by 128 threads.
    for (int idx = threadIdx.x; idx < BB * 16; idx += 128) {
        const int b = idx >> 4;
        const int c = idx & 15;
        const int j = ncol0 + c;
        const size_t grow = ((size_t)b * TT + t) * (size_t)G4H;

        const float gi = gbuf[0][b][c] + gx[grow + 0 * HH + j];
        const float gf = gbuf[1][b][c] + gx[grow + 1 * HH + j];
        const float gg = gbuf[2][b][c] + gx[grow + 2 * HH + j];
        const float go = gbuf[3][b][c] + gx[grow + 3 * HH + j];

        const float si = dev_sigmoid(gi);
        const float sf = dev_sigmoid(gf);
        const float tg = tanhf(gg);
        const float so = dev_sigmoid(go);

        const size_t sidx = (size_t)b * HH + j;
        const float cc = sf * c_state[sidx] + si * tg;
        c_state[sidx] = cc;
        const float hv = so * tanhf(cc);

        h_next[sidx] = (__bf16)hv;
        hs[((size_t)b * TT + t) * (size_t)HH + j] = (__bf16)hv;
    }
}

// ---------------------------------------------------------------------------
// Helpers: casts, bias sum, embedding gather, state init
// ---------------------------------------------------------------------------
__global__ void cast_f32_bf16(const float* __restrict__ s, __bf16* __restrict__ d,
                              long n) {
    long i = (long)blockIdx.x * blockDim.x + threadIdx.x;
    const long stride = (long)gridDim.x * blockDim.x;
    for (; i < n; i += stride) d[i] = (__bf16)s[i];
}

__global__ void bias_sum(const float* __restrict__ a, const float* __restrict__ b,
                         float* __restrict__ o, int n) {
    const int i = blockIdx.x * blockDim.x + threadIdx.x;
    if (i < n) o[i] = a[i] + b[i];
}

// One block per (b,t): gather embedding row for teacher-forced token.
__global__ void embed_tokens(const int* __restrict__ target,
                             const __bf16* __restrict__ emb,
                             __bf16* __restrict__ x) {
    const int m = blockIdx.x;          // b*T + t
    const int t = m & (TT - 1);
    const int b = m >> 6;
    const int tok = (t == 0) ? 1 /*START*/ : target[b * TT + t - 1];
    const __bf16* src = emb + (size_t)tok * HH;
    __bf16* dst = x + (size_t)m * HH;
    for (int j = threadIdx.x; j < HH; j += blockDim.x) dst[j] = src[j];
}

__global__ void init_state(const float* __restrict__ hidden,
                           float* __restrict__ c_state,
                           __bf16* __restrict__ h0) {
    const int i = blockIdx.x * blockDim.x + threadIdx.x;   // 32768 elems
    const float v = hidden[i];
    c_state[i] = v;
    h0[i] = (__bf16)v;
}

// ---------------------------------------------------------------------------
// Workspace layout (bytes, 256-aligned)
// ---------------------------------------------------------------------------
static constexpr size_t OFF_EMB   = 0;                                   // V*H bf16
static constexpr size_t SZ_EMB    = (size_t)VV * HH * 2;                 // 65,536,000
static constexpr size_t OFF_WIH   = OFF_EMB + SZ_EMB;
static constexpr size_t SZ_W      = (size_t)G4H * HH * 2;                // 8,388,608
static constexpr size_t OFF_WHH   = OFF_WIH + SZ_W;
static constexpr size_t OFF_XBF   = OFF_WHH + SZ_W;
static constexpr size_t SZ_XBF    = (size_t)MROWS * HH * 2;              // 4,194,304
static constexpr size_t OFF_HSBF  = OFF_XBF + SZ_XBF;
static constexpr size_t OFF_GX    = OFF_HSBF + SZ_XBF;
static constexpr size_t SZ_GX     = (size_t)MROWS * G4H * 4;             // 33,554,432
static constexpr size_t OFF_HBUF0 = OFF_GX + SZ_GX;
static constexpr size_t SZ_HBUF   = (size_t)BB * HH * 2;                 // 65,536
static constexpr size_t OFF_HBUF1 = OFF_HBUF0 + SZ_HBUF;
static constexpr size_t OFF_CST   = OFF_HBUF1 + SZ_HBUF;
static constexpr size_t SZ_CST    = (size_t)BB * HH * 4;                 // 131,072
static constexpr size_t OFF_BSUM  = OFF_CST + SZ_CST;

extern "C" void kernel_launch(void* const* d_in, const int* in_sizes, int n_in,
                              void* d_out, int out_size, void* d_ws, size_t ws_size,
                              hipStream_t stream) {
    const float* hidden    = (const float*)d_in[0];   // [B,H]
    const int*   target    = (const int*)  d_in[1];   // [B,T]
    const float* embedding = (const float*)d_in[2];   // [V,H]
    const float* W_ih      = (const float*)d_in[3];   // [4H,H]
    const float* W_hh      = (const float*)d_in[4];   // [4H,H]
    const float* b_ih      = (const float*)d_in[5];   // [4H]
    const float* b_hh      = (const float*)d_in[6];   // [4H]
    const float* out_bias  = (const float*)d_in[7];   // [V]
    float* out = (float*)d_out;                       // [B,T,V] f32

    char* ws = (char*)d_ws;
    __bf16* emb_bf = (__bf16*)(ws + OFF_EMB);
    __bf16* wih_bf = (__bf16*)(ws + OFF_WIH);
    __bf16* whh_bf = (__bf16*)(ws + OFF_WHH);
    __bf16* x_bf   = (__bf16*)(ws + OFF_XBF);
    __bf16* hs_bf  = (__bf16*)(ws + OFF_HSBF);
    float*  gx     = (float*) (ws + OFF_GX);
    __bf16* hbuf[2] = { (__bf16*)(ws + OFF_HBUF0), (__bf16*)(ws + OFF_HBUF1) };
    float*  c_st   = (float*) (ws + OFF_CST);
    float*  bsum   = (float*) (ws + OFF_BSUM);

    // 1) Cast weights & embedding to bf16 (embedding bf16 = 64 MB -> L2-resident)
    cast_f32_bf16<<<2048, 256, 0, stream>>>(embedding, emb_bf, (long)VV * HH);
    cast_f32_bf16<<<1024, 256, 0, stream>>>(W_ih, wih_bf, (long)G4H * HH);
    cast_f32_bf16<<<1024, 256, 0, stream>>>(W_hh, whh_bf, (long)G4H * HH);
    bias_sum<<<(G4H + 255) / 256, 256, 0, stream>>>(b_ih, b_hh, bsum, G4H);

    // 2) Teacher-forced embedding gather -> x_bf [2048,1024]
    embed_tokens<<<MROWS, 256, 0, stream>>>(target, emb_bf, x_bf);

    // 3) gx = x @ W_ih^T + (b_ih + b_hh)   [2048 x 4096], K=1024
    {
        dim3 grid(G4H / 256, MROWS / 64);
        gemm_bf16_wmma<<<grid, 256, 0, stream>>>(x_bf, wih_bf, gx, bsum,
                                                 MROWS, G4H, HH);
    }

    // 4) Init h0/c0 from `hidden`
    init_state<<<(BB * HH) / 256, 256, 0, stream>>>(hidden, c_st, hbuf[0]);

    // 5) Recurrent scan: 64 sequential steps (h double-buffered in bf16)
    for (int t = 0; t < TT; ++t) {
        lstm_step<<<HH / 16, 128, 0, stream>>>(hbuf[t & 1], whh_bf, gx, c_st,
                                               hbuf[(t + 1) & 1], hs_bf, t);
    }

    // 6) logits = hs @ embedding^T + out_bias   [2048 x 32000], K=1024
    {
        dim3 grid(VV / 256, MROWS / 64);   // 125 x 32
        gemm_bf16_wmma<<<grid, 256, 0, stream>>>(hs_bf, emb_bf, out, out_bias,
                                                 MROWS, VV, HH);
    }
}
